// EntropyLinear_24635932410290
// MI455X (gfx1250) — matrix operand order repaired and checked by reading, not compiled, not run on hardware
//
#include <hip/hip_runtime.h>
#include <hip/hip_bf16.h>

typedef _Float16 v16h __attribute__((ext_vector_type(16)));
typedef _Float16 h8   __attribute__((ext_vector_type(8)));
typedef float    v8f  __attribute__((ext_vector_type(8)));

#define NCLASS 50
#define IN_DIM 256
#define OUT_DIM 32
#define TILE_M 128
#define TEMP_INV (1.0f / 0.6f)

// ---------------------------------------------------------------------------
// Phase 1: gamma = sum_o |W[c,o,i]|; alpha_norm = exp((gamma - max)/T)
// (softmax denominator cancels against the /max(alpha) normalization)
// Wg[c,o,i] = (f16)(W[c,o,i] * alpha_norm[c,i])
// ---------------------------------------------------------------------------
__global__ __launch_bounds__(IN_DIM)
void entropy_prep_kernel(const float* __restrict__ W, _Float16* __restrict__ Wg) {
    const int c = blockIdx.x;       // class
    const int i = threadIdx.x;      // input-feature column 0..255

    const float* Wc = W + (size_t)c * OUT_DIM * IN_DIM;
    float w[OUT_DIM];
    float gamma = 0.0f;
#pragma unroll
    for (int o = 0; o < OUT_DIM; ++o) {
        w[o] = Wc[o * IN_DIM + i];          // coalesced over i
        gamma += fabsf(w[o]);
    }

    __shared__ float red[IN_DIM];
    red[i] = gamma;
    __syncthreads();
#pragma unroll
    for (int s = IN_DIM / 2; s > 0; s >>= 1) {
        if (i < s) red[i] = fmaxf(red[i], red[i + s]);
        __syncthreads();
    }
    const float gmax  = red[0];
    const float scale = expf((gamma - gmax) * TEMP_INV);   // == alpha_norm[c,i]

    _Float16* Wgc = Wg + (size_t)c * OUT_DIM * IN_DIM;
#pragma unroll
    for (int o = 0; o < OUT_DIM; ++o)
        Wgc[o * IN_DIM + i] = (_Float16)(w[o] * scale);    // coalesced over i
}

// ---------------------------------------------------------------------------
// Phase 2: per-class GEMM via V_WMMA_F32_16X16X32_F16.
// Block: 256 threads = 8 waves; each wave owns a 16-row M-tile of the
// 128-row x tile staged in LDS as f16. A-fragments (K=256 -> 8 chunks of 32)
// are held in registers and reused across all 50 classes.
// ---------------------------------------------------------------------------
__global__ __launch_bounds__(256)
void entropy_gemm_kernel(const float* __restrict__ x,
                         const _Float16* __restrict__ Wg,
                         const float* __restrict__ bias,
                         float* __restrict__ out) {
    __shared__ __align__(32) _Float16 xs[TILE_M * IN_DIM];   // 64 KB

    const int t         = threadIdx.x;
    const int block_row = blockIdx.x * TILE_M;

    // ---- stage x tile to LDS, fp32 -> f16 (coalesced b128 loads) ----------
    const float4* xv = (const float4*)(x + (size_t)block_row * IN_DIM);
#pragma unroll
    for (int it = 0; it < (TILE_M * IN_DIM / 4) / 256; ++it) {
        const int idx4 = t + it * 256;
        const float4 v = xv[idx4];
        const int b = idx4 * 4;
        xs[b + 0] = (_Float16)v.x;
        xs[b + 1] = (_Float16)v.y;
        xs[b + 2] = (_Float16)v.z;
        xs[b + 3] = (_Float16)v.w;
    }
    __syncthreads();

    const int wave = t >> 5;
    const int lane = t & 31;
    const int m    = lane & 15;    // A row within 16-row tile / D column id
    const int hi   = lane >> 4;    // lane half selects K sub-groups
    const int row  = wave * 16 + m;

    // ---- preload the 8 A-fragments (16x32 f16, contiguous K order per lane half) ---------
    // halves 0..7  -> K = kc*32 + 16*hi + j
    // halves 8..15 -> K = kc*32 + 16*hi + 8 + j
    union Af { v16h v; h8 h[2]; };
    Af a[8];
#pragma unroll
    for (int kc = 0; kc < 8; ++kc) {
        const _Float16* p = &xs[row * IN_DIM + kc * 32 + hi * 16];
        a[kc].h[0] = *(const h8*)(p);
        a[kc].h[1] = *(const h8*)(p + 8);
    }

    // B layout (32x16 f16): lane holds column n = lane%16,
    // halves = 16 contiguous K starting at kc*32 + 16*hi.
    const int n = m;
    const size_t class_stride = (size_t)OUT_DIM * IN_DIM;   // 8192 halves
    const _Float16* Wbase = Wg + (size_t)n * IN_DIM + (size_t)hi * 16;

    for (int c = 0; c < NCLASS; ++c) {
        const _Float16* Wc = Wbase + (size_t)c * class_stride;
        if (c + 1 < NCLASS)
            __builtin_prefetch(Wc + class_stride, 0, 3);    // global_prefetch

        v8f acc0 = {0.f, 0.f, 0.f, 0.f, 0.f, 0.f, 0.f, 0.f};
        v8f acc1 = acc0;
#pragma unroll
        for (int kc = 0; kc < 8; ++kc) {
            const v16h b0 = *(const v16h*)(Wc + kc * 32);                 // o in [0,16)
            const v16h b1 = *(const v16h*)(Wc + 16 * IN_DIM + kc * 32);   // o in [16,32)
            acc0 = __builtin_amdgcn_wmma_f32_16x16x32_f16(
                       false, a[kc].v, false, b0, (short)0, acc0, false, false);
            acc1 = __builtin_amdgcn_wmma_f32_16x16x32_f16(
                       false, a[kc].v, false, b1, (short)0, acc1, false, false);
        }

        const float b0s = bias[c * OUT_DIM + n];
        const float b1s = bias[c * OUT_DIM + n + 16];

        // D layout: VGPR r -> M = r + 8*hi, N = lane%16
        const size_t obase = ((size_t)block_row + wave * 16 + hi * 8) *
                             (NCLASS * OUT_DIM) + (size_t)c * OUT_DIM + n;
#pragma unroll
        for (int r = 0; r < 8; ++r) {
            out[obase + (size_t)r * (NCLASS * OUT_DIM)]      = acc0[r] + b0s;
            out[obase + (size_t)r * (NCLASS * OUT_DIM) + 16] = acc1[r] + b1s;
        }
    }
}

extern "C" void kernel_launch(void* const* d_in, const int* in_sizes, int n_in,
                              void* d_out, int out_size, void* d_ws, size_t ws_size,
                              hipStream_t stream) {
    const float* x    = (const float*)d_in[0];   // [8192, 256]
    const float* W    = (const float*)d_in[1];   // [50, 32, 256]
    const float* bias = (const float*)d_in[2];   // [50, 1, 32]
    float* out        = (float*)d_out;           // [8192, 50, 32]

    _Float16* Wg = (_Float16*)d_ws;              // [50, 32, 256] f16 = 0.8 MB

    entropy_prep_kernel<<<NCLASS, IN_DIM, 0, stream>>>(W, Wg);
    entropy_gemm_kernel<<<(8192 / TILE_M), 256, 0, stream>>>(x, Wg, bias, out);
}
